// APConv_13915694039582
// MI455X (gfx1250) — matrix operand order repaired and checked by reading, not compiled, not run on hardware
//
#include <hip/hip_runtime.h>
#include <hip/hip_bf16.h>

typedef __attribute__((ext_vector_type(16))) __bf16 v16bf;
typedef __attribute__((ext_vector_type(4)))  __bf16 bf4;
typedef __attribute__((ext_vector_type(8)))  float  v8f;
typedef unsigned short ushort_t;

union Frag {
    v16bf v;
    int4  q[2];
};

__device__ __forceinline__ ushort_t f2bf(float f) {
    return __builtin_bit_cast(ushort_t, (__bf16)f);
}

// pack 4 f32 -> 4 bf16 (2x v_cvt_pk_bf16_f32) as a uint2
__device__ __forceinline__ uint2 f4_to_bf4(float4 v) {
    bf4 b = { (__bf16)v.x, (__bf16)v.y, (__bf16)v.z, (__bf16)v.w };
    return __builtin_bit_cast(uint2, b);
}

// ---------------------------------------------------------------------------
// Zero the aggregation buffer (atomically accumulated each call).
// ---------------------------------------------------------------------------
__global__ void apconv_zero_kernel(float* p, int n) {
    int i = blockIdx.x * blockDim.x + threadIdx.x;
    if (i < n) p[i] = 0.0f;
}

// ---------------------------------------------------------------------------
// Pack an f32 KxN weight matrix into bf16 WMMA B-fragment layout.
// Fragment f = kt*ntiles+nt holds 32 lanes x 16 bf16; element i of lane L is
// W[kt*32 + (i>>3)*16 + (L>>4)*8 + (i&7)][nt*16 + (L&15)]  (ISA 7.12.2 16-bit B layout).
// ---------------------------------------------------------------------------
__global__ void apconv_pack_kernel(const float* __restrict__ src, int K, int N,
                                   int ntiles, ushort_t* __restrict__ dst) {
    int g = blockIdx.x * blockDim.x + threadIdx.x;
    if (g >= K * N) return;
    int f    = g >> 9;
    int r    = g & 511;
    int lane = r >> 4;
    int i    = r & 15;
    int kt   = f / ntiles;
    int nt   = f - kt * ntiles;
    int Krow = kt * 32 + (i >> 3) * 16 + (lane >> 4) * 8 + (i & 7);
    int Ncol = nt * 16 + (lane & 15);
    dst[g] = f2bf(src[(size_t)Krow * N + Ncol]);
}

// ---------------------------------------------------------------------------
// Edge MLP + segment-sum scatter.
// Each wave: 32 edges (two 16-row M-tiles -> every B fragment feeds 2 WMMAs).
// msg = concat(edge_hid, ue_hid[src]) : 32x128 bf16.
// h = relu(msg@W1a+b1a)@W1b + b1b, then atomicAdd into agg[dst].
// ---------------------------------------------------------------------------
__global__ __launch_bounds__(128) void apconv_edge_kernel(
    const float* __restrict__ edge_hid, const float* __restrict__ ue_hid,
    const int* __restrict__ srcIdx, const int* __restrict__ dstIdx,
    const ushort_t* __restrict__ pW1a, const ushort_t* __restrict__ pW1b,
    const float* __restrict__ b1a, const float* __restrict__ b1b,
    float* __restrict__ agg, int nE, int nTiles) {

    __shared__ ushort_t wlds[32768];     // W1a [0..16383], W1b [16384..32767]
    __shared__ ushort_t msg[4][4096];    // per-wave 32x128 bf16
    __shared__ ushort_t h1 [4][4096];    // per-wave 32x128 bf16

    {   // cooperative weight load: 64KB = 4096 int4
        const int4* s1 = (const int4*)pW1a;
        const int4* s2 = (const int4*)pW1b;
        int4* dW = (int4*)wlds;
        for (int t = threadIdx.x; t < 2048; t += 128) dW[t] = s1[t];
        for (int t = threadIdx.x; t < 2048; t += 128) dW[2048 + t] = s2[t];
    }
    __syncthreads();

    const int lane = threadIdx.x & 31;
    const int wv   = threadIdx.x >> 5;
    const int half = lane >> 4;
    const int nlo  = lane & 15;
    ushort_t* msgw = msg[wv];
    ushort_t* h1w  = h1[wv];

    float bias1[8], bias2[8];
#pragma unroll
    for (int nt = 0; nt < 8; ++nt) {
        bias1[nt] = b1a[nt * 16 + nlo];
        bias2[nt] = b1b[nt * 16 + nlo];
    }

    // invariant per-lane bases (immediate-offset friendly)
    ushort_t* h1st0 = &h1w[(half * 8) * 128 + nlo];        // tile0: + i*128 + nt*16
    ushort_t* h1st1 = &h1w[(16 + half * 8) * 128 + nlo];   // tile1
    const ushort_t* aA1_0 = &msgw[nlo * 128 + half * 8];
    const ushort_t* aA1_1 = &msgw[(nlo + 16) * 128 + half * 8];
    const ushort_t* aA2_0 = &h1w [nlo * 128 + half * 8];
    const ushort_t* aA2_1 = &h1w [(nlo + 16) * 128 + half * 8];
    const ushort_t* wB1 = &wlds[lane * 16];
    const ushort_t* wB2 = &wlds[16384 + lane * 16];
    float* aggL = agg + nlo;

    for (int tile = blockIdx.x; tile < nTiles; tile += gridDim.x) {
        const int eBase = tile * 128 + wv * 32;

        // ---- stage msg tile (gather + f32->bf16) --------------------------
        // row k, lanes 0-15 -> edge_hid cols, lanes 16-31 -> ue_hid gather
#pragma unroll
        for (int k = 0; k < 32; ++k) {
            int c4 = lane * 4;                 // 0..124
            int e  = eBase + k; if (e >= nE) e = nE - 1;
            const float* p = (c4 < 64)
                ? edge_hid + (size_t)e * 64 + c4
                : ue_hid   + (size_t)srcIdx[e] * 64 + (c4 - 64);
            float4 v = *(const float4*)p;
            *(uint2*)&msgw[k * 128 + c4] = f4_to_bf4(v);
        }
        __builtin_amdgcn_wave_barrier();

        // ---- layer 1: msg(32x128) @ W1a(128x128), bias + relu -------------
        {
            v8f acc0[8] = {}, acc1[8] = {};
#pragma unroll
            for (int kt = 0; kt < 4; ++kt) {
                Frag a0, a1;
                a0.q[0] = *(const int4*)(aA1_0 + kt * 32);
                a0.q[1] = *(const int4*)(aA1_0 + kt * 32 + 16);
                a1.q[0] = *(const int4*)(aA1_1 + kt * 32);
                a1.q[1] = *(const int4*)(aA1_1 + kt * 32 + 16);
#pragma unroll
                for (int nt = 0; nt < 8; ++nt) {
                    Frag b;
                    const ushort_t* pb = wB1 + (kt * 8 + nt) * 512;
                    b.q[0] = *(const int4*)pb;
                    b.q[1] = *(const int4*)(pb + 8);
                    acc0[nt] = __builtin_amdgcn_wmma_f32_16x16x32_bf16(
                        false, a0.v, false, b.v, (short)0, acc0[nt], false, false);
                    acc1[nt] = __builtin_amdgcn_wmma_f32_16x16x32_bf16(
                        false, a1.v, false, b.v, (short)0, acc1[nt], false, false);
                }
            }
#pragma unroll
            for (int nt = 0; nt < 8; ++nt) {
                float bs = bias1[nt];
#pragma unroll
                for (int i = 0; i < 8; ++i) {
                    h1st0[i * 128 + nt * 16] = f2bf(fmaxf(acc0[nt][i] + bs, 0.0f));
                    h1st1[i * 128 + nt * 16] = f2bf(fmaxf(acc1[nt][i] + bs, 0.0f));
                }
            }
        }
        __builtin_amdgcn_wave_barrier();

        // ---- layer 2: h1(32x128) @ W1b(128x128), bias, atomic scatter -----
        {
            int  aoff[16];      // agg dword offsets (excl. nlo, added via aggL) per owned row
            bool dval[16];
#pragma unroll
            for (int j = 0; j < 16; ++j) {
                // j<8: tile0 rows half*8+i ; j>=8: tile1 rows 16+half*8+i
                int e   = eBase + (j < 8 ? 0 : 16) + half * 8 + (j & 7);
                dval[j] = e < nE;
                int d   = dstIdx[dval[j] ? e : (nE - 1)];
                aoff[j] = d * 128;
            }
            v8f acc0[8] = {}, acc1[8] = {};
#pragma unroll
            for (int kt = 0; kt < 4; ++kt) {
                Frag a0, a1;
                a0.q[0] = *(const int4*)(aA2_0 + kt * 32);
                a0.q[1] = *(const int4*)(aA2_0 + kt * 32 + 16);
                a1.q[0] = *(const int4*)(aA2_1 + kt * 32);
                a1.q[1] = *(const int4*)(aA2_1 + kt * 32 + 16);
#pragma unroll
                for (int nt = 0; nt < 8; ++nt) {
                    Frag b;
                    const ushort_t* pb = wB2 + (kt * 8 + nt) * 512;
                    b.q[0] = *(const int4*)pb;
                    b.q[1] = *(const int4*)(pb + 8);
                    acc0[nt] = __builtin_amdgcn_wmma_f32_16x16x32_bf16(
                        false, a0.v, false, b.v, (short)0, acc0[nt], false, false);
                    acc1[nt] = __builtin_amdgcn_wmma_f32_16x16x32_bf16(
                        false, a1.v, false, b.v, (short)0, acc1[nt], false, false);
                }
            }
#pragma unroll
            for (int nt = 0; nt < 8; ++nt) {
                float bs = bias2[nt];
#pragma unroll
                for (int i = 0; i < 8; ++i) {
                    if (dval[i])
                        unsafeAtomicAdd(aggL + aoff[i] + nt * 16, acc0[nt][i] + bs);
                    if (dval[8 + i])
                        unsafeAtomicAdd(aggL + aoff[8 + i] + nt * 16, acc1[nt][i] + bs);
                }
            }
        }
        __builtin_amdgcn_wave_barrier();
    }
}

// ---------------------------------------------------------------------------
// AP MLP: new_ap = relu(concat(ap_hid, agg) @ W2a + b2a) @ W2b + b2b
// Each wave: 16 APs. z tile is 16x192 bf16.
// ---------------------------------------------------------------------------
__global__ __launch_bounds__(128) void apconv_ap_kernel(
    const float* __restrict__ ap_hid, const float* __restrict__ agg,
    const ushort_t* __restrict__ pW2a, const ushort_t* __restrict__ pW2b,
    const float* __restrict__ b2a, const float* __restrict__ b2b,
    float* __restrict__ out, int nAP) {

    __shared__ ushort_t wlds[32768];     // W2a [0..24575], W2b [24576..32767]
    __shared__ ushort_t zt[4][3072];     // per-wave 16x192 bf16
    __shared__ ushort_t h2[4][2048];     // per-wave 16x128 bf16

    {   // cooperative weight load: 48KB + 16KB
        const int4* s1 = (const int4*)pW2a;   // 3072 int4
        const int4* s2 = (const int4*)pW2b;   // 1024 int4
        int4* dW = (int4*)wlds;
        for (int t = threadIdx.x; t < 3072; t += 128) dW[t] = s1[t];
        for (int t = threadIdx.x; t < 1024; t += 128) dW[3072 + t] = s2[t];
    }
    __syncthreads();

    const int lane = threadIdx.x & 31;
    const int wv   = threadIdx.x >> 5;
    const int half = lane >> 4;
    const int nlo  = lane & 15;
    ushort_t* zw = zt[wv];
    ushort_t* hw = h2[wv];
    const int apBase = blockIdx.x * 64 + wv * 16;

    float biasA[8], biasB[4];
#pragma unroll
    for (int nt = 0; nt < 8; ++nt) biasA[nt] = b2a[nt * 16 + nlo];
#pragma unroll
    for (int nt = 0; nt < 4; ++nt) biasB[nt] = b2b[nt * 16 + nlo];

    // ---- stage z tile: 16 rows x 192 cols (ap_hid | agg) -------------------
#pragma unroll
    for (int it = 0; it < 24; ++it) {
        int idx = it * 32 + lane;
        int row = idx / 48;
        int c4  = (idx % 48) * 4;
        int a   = apBase + row; if (a >= nAP) a = nAP - 1;
        const float* p = (c4 < 64)
            ? ap_hid + (size_t)a * 64 + c4
            : agg    + (size_t)a * 128 + (c4 - 64);
        float4 v = *(const float4*)p;
        *(uint2*)&zw[row * 192 + c4] = f4_to_bf4(v);
    }
    __builtin_amdgcn_wave_barrier();

    ushort_t* h2st = &hw[(half * 8) * 128 + nlo];
    const ushort_t* aA = &zw[nlo * 192 + half * 8];
    const ushort_t* aB = &hw[nlo * 128 + half * 8];

    // ---- layer A: z(16x192) @ W2a(192x128), bias + relu --------------------
    {
        v8f acc[8] = {};
#pragma unroll
        for (int kt = 0; kt < 6; ++kt) {
            Frag a;
            a.q[0] = *(const int4*)(aA + kt * 32);
            a.q[1] = *(const int4*)(aA + kt * 32 + 16);
#pragma unroll
            for (int nt = 0; nt < 8; ++nt) {
                Frag b;
                const ushort_t* pb = &wlds[(kt * 8 + nt) * 512 + lane * 16];
                b.q[0] = *(const int4*)pb;
                b.q[1] = *(const int4*)(pb + 8);
                acc[nt] = __builtin_amdgcn_wmma_f32_16x16x32_bf16(
                    false, a.v, false, b.v, (short)0, acc[nt], false, false);
            }
        }
#pragma unroll
        for (int nt = 0; nt < 8; ++nt) {
            float bs = biasA[nt];
#pragma unroll
            for (int i = 0; i < 8; ++i)
                h2st[i * 128 + nt * 16] = f2bf(fmaxf(acc[nt][i] + bs, 0.0f));
        }
    }
    __builtin_amdgcn_wave_barrier();

    // ---- layer B: h2(16x128) @ W2b(128x64), bias, store --------------------
    {
        float* obase[8];
        bool   oval[8];
#pragma unroll
        for (int i = 0; i < 8; ++i) {
            int a_ = apBase + i + half * 8;
            oval[i]  = a_ < nAP;
            obase[i] = out + (size_t)(oval[i] ? a_ : 0) * 64 + nlo;
        }
        v8f acc[4] = {};
#pragma unroll
        for (int kt = 0; kt < 4; ++kt) {
            Frag a;
            a.q[0] = *(const int4*)(aB + kt * 32);
            a.q[1] = *(const int4*)(aB + kt * 32 + 16);
#pragma unroll
            for (int nt = 0; nt < 4; ++nt) {
                Frag b;
                const ushort_t* pb = &wlds[24576 + (kt * 4 + nt) * 512 + lane * 16];
                b.q[0] = *(const int4*)pb;
                b.q[1] = *(const int4*)(pb + 8);
                acc[nt] = __builtin_amdgcn_wmma_f32_16x16x32_bf16(
                    false, a.v, false, b.v, (short)0, acc[nt], false, false);
            }
        }
#pragma unroll
        for (int nt = 0; nt < 4; ++nt) {
            float bs = biasB[nt];
#pragma unroll
            for (int i = 0; i < 8; ++i) {
                if (oval[i]) obase[i][nt * 16] = acc[nt][i] + bs;
            }
        }
    }
}

// ---------------------------------------------------------------------------
extern "C" void kernel_launch(void* const* d_in, const int* in_sizes, int n_in,
                              void* d_out, int out_size, void* d_ws, size_t ws_size,
                              hipStream_t stream) {
    const float* ue_hid   = (const float*)d_in[0];
    const float* ap_hid   = (const float*)d_in[1];
    const float* edge_hid = (const float*)d_in[2];
    const int*   src      = (const int*)d_in[3];
    const int*   dst      = (const int*)d_in[4];
    const float* W1a      = (const float*)d_in[5];
    const float* b1a      = (const float*)d_in[6];
    const float* W1b      = (const float*)d_in[7];
    const float* b1b      = (const float*)d_in[8];
    const float* W2a      = (const float*)d_in[9];
    const float* b2a      = (const float*)d_in[10];
    const float* W2b      = (const float*)d_in[11];
    const float* b2b      = (const float*)d_in[12];

    const int D = 64, H = 128;
    const int nAP = in_sizes[1] / D;
    const int nE  = in_sizes[3];

    // workspace layout
    float*    agg  = (float*)d_ws;
    size_t    aggB = (size_t)nAP * H * sizeof(float);
    ushort_t* pW1a = (ushort_t*)((char*)d_ws + aggB);   // 2D x H   = 16384
    ushort_t* pW1b = pW1a + 16384;                      // H  x H   = 16384
    ushort_t* pW2a = pW1b + 16384;                      // (D+H)x H = 24576
    ushort_t* pW2b = pW2a + 24576;                      // H  x D   =  8192

    // 1) zero aggregation buffer
    {
        int n = nAP * H;
        apconv_zero_kernel<<<(n + 255) / 256, 256, 0, stream>>>(agg, n);
    }
    // 2) pack weights into bf16 WMMA B-fragment layout
    apconv_pack_kernel<<<(16384 + 255) / 256, 256, 0, stream>>>(W1a, 2 * D, H, 8, pW1a);
    apconv_pack_kernel<<<(16384 + 255) / 256, 256, 0, stream>>>(W1b, H, H, 8, pW1b);
    apconv_pack_kernel<<<(24576 + 255) / 256, 256, 0, stream>>>(W2a, D + H, H, 8, pW2a);
    apconv_pack_kernel<<<( 8192 + 255) / 256, 256, 0, stream>>>(W2b, H, D, 4, pW2b);

    // 3) edge MLP + segment-sum (the 105-GFLOP hot loop), 128 edges/block-tile
    {
        int nTiles = (nE + 127) / 128;
        int blocks = nTiles < 2500 ? nTiles : 2500;
        apconv_edge_kernel<<<blocks, 128, 0, stream>>>(
            edge_hid, ue_hid, src, dst, pW1a, pW1b, b1a, b1b, agg, nE, nTiles);
    }
    // 4) AP node MLP
    apconv_ap_kernel<<<(nAP + 63) / 64, 128, 0, stream>>>(
        ap_hid, agg, pW2a, pW2b, b2a, b2b, (float*)d_out, nAP);
}
